// ChildSumTreeLSTMCell_64622077935700
// MI455X (gfx1250) — compile-verified, matrix operand order checked by reading, not run on hardware
//
#include <hip/hip_runtime.h>
#include <hip/hip_bf16.h>
#include <math.h>

#define NN 16384
#define KK 16
#define DD 128
#define HH 128
#define EE 259
#define EPAD 288      // K dim padded to multiple of 32
#define OPAD 272      // e1 output rows padded to multiple of 16
#define ROWS (NN * KK)
#define THREADS 128   // 4 waves (wave32)
#define NWAVES 4

typedef __bf16 bf16_t;
typedef __bf16 v16bf __attribute__((ext_vector_type(16)));
typedef float  v8f   __attribute__((ext_vector_type(8)));

__device__ __forceinline__ v8f wmma_bf16(v16bf a, v16bf b, v8f c) {
  // D = A(16x32 bf16) * B(32x16 bf16) + C(16x16 f32)
  return __builtin_amdgcn_wmma_f32_16x16x32_bf16(false, a, false, b, (short)0, c,
                                                 false, false);
}

union Frag { uint4 q[2]; v16bf v; };
union BQ   { bf16_t b[4]; uint2 u; };   // 4 packed bf16 <-> 8 bytes

__device__ __forceinline__ uint2 pack4(float x, float y, float z, float w) {
  BQ t;
  t.b[0] = (bf16_t)x; t.b[1] = (bf16_t)y; t.b[2] = (bf16_t)z; t.b[3] = (bf16_t)w;
  return t.u;
}

// Load a 16x32 bf16 WMMA A/B fragment from a row-major tile.
// Lane holds row (lane&15); its 16 elements are K = half*8+{0..7} and 16+half*8+{0..7}.
__device__ __forceinline__ v16bf ld_frag_bf16(const bf16_t* tile, int stride, int kbase) {
  const int lane = threadIdx.x & 31;
  const int m = lane & 15, half = lane >> 4;
  const bf16_t* p = tile + (size_t)m * stride + kbase + half * 8;
  Frag f;
  f.q[0] = *reinterpret_cast<const uint4*>(p);
  f.q[1] = *reinterpret_cast<const uint4*>(p + 16);
  return f.v;
}

// One 16-row x 16-col output tile: KF K-steps of 16x16x32 WMMA.
// B fragments are explicitly double-buffered (load K-step j+1 before the WMMA
// of K-step j) so >=2 ds_load_b128 stay in flight under every WMMA: the
// s_wait_dscnt before each WMMA becomes partial instead of a full LDS stall.
template <int KF>
__device__ __forceinline__ v8f gemm_tile(const v16bf* a, const bf16_t* bt, int stride) {
  v8f c = {};
  v16bf b = ld_frag_bf16(bt, stride, 0);
#pragma unroll
  for (int j = 0; j < KF; ++j) {
    v16bf bn = b;
    if (j + 1 < KF) bn = ld_frag_bf16(bt, stride, (j + 1) * 32);
    c = wmma_bf16(a[j], b, c);
    b = bn;
  }
  return c;
}

// ---------------------------------------------------------------------------
// Kernel A: fused 2-layer edge MLP + h*edge_w, writes h2 = [hw | embed] (bf16)
// LDS: e1 [272][288] bf16 + e2 [128][288] bf16 + biases + per-wave x/y1 tiles
// xt region is reused to stage the h tile (f32); y1 region is reused as the
// hw output tile so all global traffic is coalesced 128b.
// ---------------------------------------------------------------------------
__global__ void __launch_bounds__(THREADS) edge_mlp_kernel(
    const float* __restrict__ h, const float* __restrict__ src_embed,
    const float* __restrict__ dst_embed, const float* __restrict__ edge_type,
    const float* __restrict__ embed,
    const float* __restrict__ e1_w, const float* __restrict__ e1_b,
    const float* __restrict__ e2_w, const float* __restrict__ e2_b,
    bf16_t* __restrict__ h2, int n_tiles, int total_waves) {
  extern __shared__ char smem[];
  bf16_t* e1s  = (bf16_t*)smem;                                   // [OPAD][EPAD]
  bf16_t* e2s  = (bf16_t*)(smem + (size_t)OPAD * EPAD * 2);       // [HH][EPAD]
  float*  e1bs = (float*)(smem + (size_t)OPAD * EPAD * 2 + (size_t)HH * EPAD * 2);
  float*  e2bs = e1bs + OPAD;
  bf16_t* xbase = (bf16_t*)(e2bs + HH);                           // 4 waves x [16][EPAD] x 2

  const int tid  = threadIdx.x;
  const int wave = tid >> 5;
  const int lane = tid & 31;
  const int half = lane >> 4;
  const int col  = lane & 15;

  // ---- stage weights into LDS as bf16 (zero padded); rows per wave, cols per lane
  for (int o = wave; o < OPAD; o += NWAVES)
    for (int k = lane; k < EPAD; k += 32) {
      float v = (o < EE && k < EE) ? e1_w[(size_t)o * EE + k] : 0.0f;
      e1s[(size_t)o * EPAD + k] = (bf16_t)v;
    }
  for (int o = wave; o < HH; o += NWAVES)
    for (int k = lane; k < EPAD; k += 32) {
      float v = (k < EE) ? e2_w[(size_t)o * EE + k] : 0.0f;
      e2s[(size_t)o * EPAD + k] = (bf16_t)v;
    }
  for (int i = tid; i < OPAD; i += THREADS) e1bs[i] = (i < EE) ? e1_b[i] : 0.0f;
  for (int i = tid; i < HH; i += THREADS) e2bs[i] = e2_b[i];
  __syncthreads();

  bf16_t* xt  = xbase + (size_t)wave * 2 * 16 * EPAD;   // [16][EPAD] bf16
  bf16_t* y1  = xt + 16 * EPAD;                          // [16][EPAD] bf16
  float*  xtf = (float*)xt;                              // reused: [16][128] f32 (h tile)
  bf16_t* hwt = y1;                                      // reused: [16][128] bf16 (hw tile)

  const int gwave = blockIdx.x * NWAVES + wave;
  for (int t = gwave; t < n_tiles; t += total_waves) {
    const int row0 = t * 16;

    // prefetch next tile's activation rows (deterministic grid-stride next tile)
    const int tn = t + total_waves;
    if (tn < n_tiles) {
      const size_t nr = (size_t)tn * 16 * DD + lane * 64;
      __builtin_prefetch(src_embed + nr, 0, 0);
      __builtin_prefetch(dst_embed + nr, 0, 0);
      __builtin_prefetch(h + nr, 0, 0);
      __builtin_prefetch(embed + nr, 0, 0);
    }

    // ---- build x tile = [src | dst | edge_type] as bf16 (coalesced float4 loads)
    for (int i = lane; i < 16 * 64; i += 32) {           // 256 cols / 4
      const int r = i >> 6, q = i & 63;
      const int row = row0 + r;
      const float* sp = (q < 32) ? (src_embed + (size_t)row * DD + q * 4)
                                 : (dst_embed + (size_t)row * DD + (q - 32) * 4);
      float4 v = *reinterpret_cast<const float4*>(sp);
      *reinterpret_cast<uint2*>(xt + (size_t)r * EPAD + q * 4) = pack4(v.x, v.y, v.z, v.w);
    }
    for (int i = lane; i < 16 * 32; i += 32) {           // cols 256..287
      const int r = i >> 5, k = i & 31;
      float v = (k < 3) ? edge_type[(size_t)(row0 + r) * 3 + k] : 0.0f;
      xt[(size_t)r * EPAD + 256 + k] = (bf16_t)v;
    }

    // ---- stage-1 A fragments, then immediately reuse xt to stage the h tile
    v16bf a[9];
#pragma unroll
    for (int j = 0; j < 9; ++j) a[j] = ld_frag_bf16(xt, EPAD, j * 32);
    for (int i = lane; i < 16 * 32; i += 32) {           // h tile f32, coalesced
      const int r = i >> 5, q = i & 31;
      *reinterpret_cast<float4*>(xtf + (size_t)r * DD + q * 4) =
          *reinterpret_cast<const float4*>(h + (size_t)(row0 + r) * DD + q * 4);
    }

    // ---- stage 1: y1 = relu(x @ e1^T + b1)   (17 N-tiles x 9 K-steps)
    for (int nt = 0; nt < 17; ++nt) {
      v8f c = gemm_tile<9>(a, e1s + (size_t)nt * 16 * EPAD, EPAD);
      const int o = nt * 16 + col;
      const float bo = e1bs[o];
#pragma unroll
      for (int r = 0; r < 8; ++r)
        y1[(size_t)(r + 8 * half) * EPAD + o] = (bf16_t)fmaxf(c[r] + bo, 0.0f);
    }
    for (int i = lane; i < 16 * 16; i += 32) {           // zero pad cols 272..287
      const int r = i >> 4, k2 = i & 15;
      y1[(size_t)r * EPAD + 272 + k2] = (bf16_t)0.0f;
    }

    // ---- stage 2: edge_w = y1 @ e2^T + b2 ; hw = h * edge_w
#pragma unroll
    for (int j = 0; j < 9; ++j) a[j] = ld_frag_bf16(y1, EPAD, j * 32);
    for (int nt = 0; nt < 8; ++nt) {
      v8f c = gemm_tile<9>(a, e2s + (size_t)nt * 16 * EPAD, EPAD);
      const int o = nt * 16 + col;
      const float bo = e2bs[o];
#pragma unroll
      for (int r = 0; r < 8; ++r) {
        const int m = r + 8 * half;
        hwt[(size_t)m * DD + o] = (bf16_t)(xtf[(size_t)m * DD + o] * (c[r] + bo));
      }
    }

    // ---- coalesced h2 stores: hw half from LDS tile, embed half converted inline
    for (int i = lane; i < 16 * 16; i += 32) {           // hw: 16 rows x 16 uint4
      const int r = i >> 4, q = i & 15;
      *reinterpret_cast<uint4*>(h2 + (size_t)(row0 + r) * 256 + q * 8) =
          *reinterpret_cast<const uint4*>(hwt + (size_t)r * DD + q * 8);
    }
    for (int i = lane; i < 16 * 32; i += 32) {           // embed: float4 -> 4x bf16
      const int r = i >> 5, q = i & 31;
      float4 v = *reinterpret_cast<const float4*>(embed + (size_t)(row0 + r) * DD + q * 4);
      *reinterpret_cast<uint2*>(h2 + (size_t)(row0 + r) * 256 + HH + q * 4) =
          pack4(v.x, v.y, v.z, v.w);
    }
  }
}

// ---------------------------------------------------------------------------
// Kernel B: masked nl projection + child-sum (one node n per wave; the 16-row
// tile == all K children, so h_sum is a column sum of the WMMA output tile)
// ---------------------------------------------------------------------------
__global__ void __launch_bounds__(THREADS) nl_reduce_kernel(
    const bf16_t* __restrict__ h2, const float* __restrict__ c_in,
    const float* __restrict__ mask_h, const float* __restrict__ mask_c,
    const float* __restrict__ nl_w, const float* __restrict__ nl_b,
    float* __restrict__ h_sum, float* __restrict__ csum, int total_waves) {
  extern __shared__ char smem[];
  bf16_t* nls  = (bf16_t*)smem;                      // [256][256]
  float*  nlbs = (float*)(smem + 256 * 256 * 2);     // [256]
  bf16_t* tbase = (bf16_t*)(smem + 256 * 256 * 2 + 256 * 4);  // 4 waves x [16][256]
  const int tid = threadIdx.x, wave = tid >> 5, lane = tid & 31;
  const int half = lane >> 4, col = lane & 15;

  for (int o = wave; o < 256; o += NWAVES)           // nl rows are 16B aligned
    for (int q = lane; q < 64; q += 32) {
      float4 v = *reinterpret_cast<const float4*>(nl_w + (size_t)o * 256 + q * 4);
      *reinterpret_cast<uint2*>(nls + (size_t)o * 256 + q * 4) = pack4(v.x, v.y, v.z, v.w);
    }
  for (int i = tid; i < 256; i += THREADS) nlbs[i] = nl_b[i];
  __syncthreads();

  bf16_t* ht = tbase + (size_t)wave * 16 * 256;

  const int gwave = blockIdx.x * NWAVES + wave;
  for (int n = gwave; n < NN; n += total_waves) {
    const size_t row0 = (size_t)n * KK;

    const int nn2 = n + total_waves;
    if (nn2 < NN) {
      __builtin_prefetch(h2 + (size_t)nn2 * KK * 256 + lane * 128, 0, 0);
      __builtin_prefetch(c_in + (size_t)nn2 * KK * HH + lane * 64, 0, 0);
    }

    // stage h2 tile to LDS with coalesced uint4 loads
    for (int i = lane; i < 16 * 32; i += 32) {
      const int r = i >> 5, q = i & 31;
      *reinterpret_cast<uint4*>(ht + (size_t)r * 256 + q * 8) =
          *reinterpret_cast<const uint4*>(h2 + (row0 + r) * 256 + q * 8);
    }

    v16bf a[8];
#pragma unroll
    for (int j = 0; j < 8; ++j) a[j] = ld_frag_bf16(ht, 256, j * 32);
    float maskv[8];
#pragma unroll
    for (int r = 0; r < 8; ++r) maskv[r] = mask_h[row0 + 8 * half + r];

    for (int nt = 0; nt < 16; ++nt) {
      v8f c = gemm_tile<8>(a, nls + (size_t)nt * 16 * 256, 256);
      const int o = nt * 16 + col;
      const float bo = nlbs[o];
      float s = 0.0f;
#pragma unroll
      for (int r = 0; r < 8; ++r) s += (c[r] + bo) * maskv[r];
      s += __shfl_xor(s, 16, 32);        // combine the two row halves
      if (half == 0) h_sum[(size_t)n * 256 + o] = s;
    }

    // csum = sum_k c * mask_c  (lane owns 4 columns; all loads float4 coalesced)
    float4 acc = {0.0f, 0.0f, 0.0f, 0.0f};
    for (int k = 0; k < KK; ++k) {
      const float mk = mask_c[row0 + k];
      float4 cv = *reinterpret_cast<const float4*>(c_in + (row0 + k) * HH + lane * 4);
      acc.x += cv.x * mk; acc.y += cv.y * mk; acc.z += cv.z * mk; acc.w += cv.w * mk;
    }
    *reinterpret_cast<float4*>(csum + (size_t)n * HH + lane * 4) = acc;
  }
}

// ---------------------------------------------------------------------------
// Kernel C: all four gate GEMMs (stacked 512x256 weights in LDS) + LSTM epilogue
// ---------------------------------------------------------------------------
__global__ void __launch_bounds__(THREADS) gates_kernel(
    const float* __restrict__ h_sum, const float* __restrict__ csum,
    const float* __restrict__ wf_w, const float* __restrict__ wf_b, const float* __restrict__ b_f,
    const float* __restrict__ wi_w, const float* __restrict__ wi_b, const float* __restrict__ b_i,
    const float* __restrict__ wu_w, const float* __restrict__ wu_b, const float* __restrict__ b_u,
    const float* __restrict__ wo_w, const float* __restrict__ wo_b, const float* __restrict__ b_o,
    float* __restrict__ out, int total_waves) {
  extern __shared__ char smem[];
  bf16_t* wall = (bf16_t*)smem;                      // [512][256] : f|i|u|o
  float*  bs   = (float*)(smem + 512 * 256 * 2);     // [512]
  bf16_t* tbase = (bf16_t*)(smem + 512 * 256 * 2 + 512 * 4);  // 4 waves x [16][256]
  const int tid = threadIdx.x, wave = tid >> 5, lane = tid & 31;
  const int half = lane >> 4, col = lane & 15;

  for (int o = wave; o < 512; o += NWAVES) {
    const int g = o >> 7, r = o & 127;
    const float* w = (g == 0) ? wf_w : (g == 1) ? wi_w : (g == 2) ? wu_w : wo_w;
    for (int q = lane; q < 64; q += 32) {
      float4 v = *reinterpret_cast<const float4*>(w + (size_t)r * 256 + q * 4);
      *reinterpret_cast<uint2*>(wall + (size_t)o * 256 + q * 4) = pack4(v.x, v.y, v.z, v.w);
    }
  }
  for (int i = tid; i < 512; i += THREADS) {
    const int g = i >> 7, r = i & 127;
    bs[i] = (g == 0) ? (wf_b[r] + b_f[r])
          : (g == 1) ? (wi_b[r] + b_i[r])
          : (g == 2) ? (wu_b[r] + b_u[r])
                     : (wo_b[r] + b_o[r]);
  }
  __syncthreads();

  bf16_t* hst = tbase + (size_t)wave * 16 * 256;
  float* h_new = out;
  float* c_new = out + (size_t)NN * HH;

  const int gwave = blockIdx.x * NWAVES + wave;
  for (int t = gwave; t < NN / 16; t += total_waves) {
    const int n0 = t * 16;

    const int tn = t + total_waves;
    if (tn < NN / 16) {
      __builtin_prefetch(h_sum + (size_t)tn * 16 * 256 + lane * 128, 0, 0);
      __builtin_prefetch(csum + (size_t)tn * 16 * HH + lane * 64, 0, 0);
    }

    // stage h_sum tile to LDS as bf16 (coalesced float4 loads + packed cvt)
    for (int i = lane; i < 16 * 64; i += 32) {
      const int r = i >> 6, q = i & 63;
      float4 v = *reinterpret_cast<const float4*>(h_sum + (size_t)(n0 + r) * 256 + q * 4);
      *reinterpret_cast<uint2*>(hst + (size_t)r * 256 + q * 4) = pack4(v.x, v.y, v.z, v.w);
    }

    v16bf a[8];
#pragma unroll
    for (int j = 0; j < 8; ++j) a[j] = ld_frag_bf16(hst, 256, j * 32);

    for (int ht = 0; ht < 8; ++ht) {
      // four independent double-buffered WMMA chains (one per gate)
      v8f cf = gemm_tile<8>(a, wall + (size_t)(0 * 128 + ht * 16) * 256, 256);
      v8f ci = gemm_tile<8>(a, wall + (size_t)(1 * 128 + ht * 16) * 256, 256);
      v8f cu = gemm_tile<8>(a, wall + (size_t)(2 * 128 + ht * 16) * 256, 256);
      v8f co = gemm_tile<8>(a, wall + (size_t)(3 * 128 + ht * 16) * 256, 256);
      const int oc = ht * 16 + col;
      const float bf_ = bs[oc], bi_ = bs[128 + oc], bu_ = bs[256 + oc], bo_ = bs[384 + oc];
#pragma unroll
      for (int r = 0; r < 8; ++r) {
        const int n = n0 + r + 8 * half;
        const float fg = 1.0f / (1.0f + __expf(-(cf[r] + bf_)));
        const float ig = 1.0f / (1.0f + __expf(-(ci[r] + bi_)));
        const float ug = tanhf(cu[r] + bu_);
        const float og = 1.0f / (1.0f + __expf(-(co[r] + bo_)));
        const float ct = csum[(size_t)n * HH + oc];
        const float cn = ig * ug + fg * ct;
        h_new[(size_t)n * HH + oc] = og * tanhf(cn);
        c_new[(size_t)n * HH + oc] = cn;
      }
    }
  }
}

extern "C" void kernel_launch(void* const* d_in, const int* in_sizes, int n_in,
                              void* d_out, int out_size, void* d_ws, size_t ws_size,
                              hipStream_t stream) {
  (void)in_sizes; (void)n_in; (void)out_size; (void)ws_size;
  const float* h         = (const float*)d_in[0];
  const float* c         = (const float*)d_in[1];
  const float* embed     = (const float*)d_in[2];
  const float* src_embed = (const float*)d_in[3];
  const float* dst_embed = (const float*)d_in[4];
  const float* edge_type = (const float*)d_in[5];
  const float* mask_h    = (const float*)d_in[6];
  const float* mask_c    = (const float*)d_in[7];
  const float* e1_w = (const float*)d_in[8];
  const float* e1_b = (const float*)d_in[9];
  const float* e2_w = (const float*)d_in[10];
  const float* e2_b = (const float*)d_in[11];
  const float* nl_w = (const float*)d_in[12];
  const float* nl_b = (const float*)d_in[13];
  const float* wf_w = (const float*)d_in[14];
  const float* wf_b = (const float*)d_in[15];
  const float* b_f  = (const float*)d_in[16];
  const float* wi_w = (const float*)d_in[17];
  const float* wi_b = (const float*)d_in[18];
  const float* b_i  = (const float*)d_in[19];
  const float* wu_w = (const float*)d_in[20];
  const float* wu_b = (const float*)d_in[21];
  const float* b_u  = (const float*)d_in[22];
  const float* wo_w = (const float*)d_in[23];
  const float* wo_b = (const float*)d_in[24];
  const float* b_o  = (const float*)d_in[25];

  // workspace layout
  char* ws = (char*)d_ws;
  bf16_t* h2   = (bf16_t*)ws;                                    // ROWS*256 bf16
  float*  hsum = (float*)(ws + (size_t)ROWS * 256 * 2);          // NN*256 f32
  float*  csum = hsum + (size_t)NN * 256;                        // NN*128 f32

  const int blkA = 1024;
  size_t smemA = (size_t)OPAD * EPAD * 2 + (size_t)HH * EPAD * 2 +
                 (size_t)OPAD * 4 + (size_t)HH * 4 +
                 (size_t)NWAVES * 2 * 16 * EPAD * 2;             // ~305.7 KB
  edge_mlp_kernel<<<blkA, THREADS, smemA, stream>>>(
      h, src_embed, dst_embed, edge_type, embed,
      e1_w, e1_b, e2_w, e2_b, h2, ROWS / 16, blkA * NWAVES);

  const int blkB = 1024;
  size_t smemB = (size_t)256 * 256 * 2 + 256 * 4 +
                 (size_t)NWAVES * 16 * 256 * 2;                  // ~164 KB
  nl_reduce_kernel<<<blkB, THREADS, smemB, stream>>>(
      h2, c, mask_h, mask_c, nl_w, nl_b, hsum, csum, blkB * NWAVES);

  const int blkC = 256;
  size_t smemC = (size_t)512 * 256 * 2 + 512 * 4 +
                 (size_t)NWAVES * 16 * 256 * 2;                  // ~296 KB
  gates_kernel<<<blkC, THREADS, smemC, stream>>>(
      hsum, csum, wf_w, wf_b, b_f, wi_w, wi_b, b_i,
      wu_w, wu_b, b_u, wo_w, wo_b, b_o, (float*)d_out, blkC * NWAVES);
}
